// SimpleGraphSimProxy_89996744720968
// MI455X (gfx1250) — matrix operand order repaired and checked by reading, not compiled, not run on hardware
//
#include <hip/hip_runtime.h>
#include <hip/hip_bf16.h>
#include <stdint.h>

#define N_STOPS 5000
#define EMBED   128
#define BB      16
#define RR      32
#define LL      64
#define EDGES   (RR * LL)      // 2048 edges per scenario
#define MP      5008           // 313 * 16 row-padded
#define MTILES  313
#define SSTRIDE ((size_t)MP * EMBED)   // per-scenario activation stride

typedef __attribute__((ext_vector_type(16))) __bf16 v16bf;
typedef __attribute__((ext_vector_type(8)))  float  v8f;

__device__ __forceinline__ float lrelu(float v) { return v > 0.f ? v : 0.01f * v; }

// ---------------- WMMA fragment loaders (bf16) ----------------
// A-frag (16x32, MxK): per lane two 8-element chunks at kb and kb+16
__device__ __forceinline__ v16bf load_a_frag(const __bf16* p) {
    union { v16bf v; uint4 u[2]; } t;
    t.u[0] = *reinterpret_cast<const uint4*>(p);
    t.u[1] = *reinterpret_cast<const uint4*>(p + 16);
    return t.v;
}
// B-frag (32x16, KxN as W^T rows): per lane two 8-element chunks at kb and kb+16
__device__ __forceinline__ v16bf load_b_frag(const __bf16* p) {
    union { v16bf v; uint4 u[2]; } t;
    t.u[0] = *reinterpret_cast<const uint4*>(p);
    t.u[1] = *reinterpret_cast<const uint4*>(p + 16);
    return t.v;
}

// ---------------- batched GEMM: C[s] = A[s] * W (W given as W^T bf16) ----------------
// grid.x: 313*8 tiles, grid.y: scenarios; one wave per 16x16 tile, 4 x wmma 16x16x32 bf16
__global__ __launch_bounds__(32)
void k_gemm(const __bf16* __restrict__ A, const __bf16* __restrict__ BT,
            float* __restrict__ C, size_t astride, size_t cstride) {
    const int lane = threadIdx.x;
    const int half = lane >> 4;
    const int lm   = lane & 15;
    const int tM   = blockIdx.x >> 3;   // 313 M-tiles
    const int tN   = blockIdx.x & 7;    // 8 N-tiles
    const int s    = blockIdx.y;

    const __bf16* ap = A + (size_t)s * astride + (size_t)(tM * 16 + lm) * EMBED;
    const __bf16* bp = BT + (size_t)(tN * 16 + lm) * EMBED;

    v8f acc = {};
#pragma unroll
    for (int kk = 0; kk < 4; ++kk) {
        v16bf af = load_a_frag(ap + kk * 32 + half * 8);
        v16bf bf = load_b_frag(bp + kk * 32 + half * 8);
        acc = __builtin_amdgcn_wmma_f32_16x16x32_bf16(
            false, af, false, bf, (short)0, acc, false, false);
    }

    const int n  = tN * 16 + lm;
    const int mb = tM * 16 + half * 8;
    float* cp = C + (size_t)s * cstride;
#pragma unroll
    for (int r = 0; r < 8; ++r)
        cp[(size_t)(mb + r) * EMBED + n] = acc[r];
}

// ---------------- elementwise / conversion kernels ----------------
__global__ void k_node(const float* __restrict__ w_emb, const float* __restrict__ b_emb,
                       __bf16* __restrict__ node_bf) {
    int idx = blockIdx.x * 256 + threadIdx.x;
    if (idx >= MP * EMBED) return;
    int n = idx >> 7, d = idx & 127;
    float v = (n < N_STOPS) ? (w_emb[idx] + b_emb[d]) : 0.f;
    node_bf[idx] = (__bf16)v;
}

// Wt[l][n][k] = gcn_w[l][k][n] as bf16
__global__ void k_wconv(const float* __restrict__ gw, __bf16* __restrict__ wt) {
    int idx = blockIdx.x * 256 + threadIdx.x;
    if (idx >= 3 * EMBED * EMBED) return;
    int l = idx >> 14, rem = idx & 16383, n = rem >> 7, k = rem & 127;
    wt[idx] = (__bf16)gw[(l << 14) + (k << 7) + n];
}

// ---------------- degrees, all scenarios at once ----------------
__global__ void k_deg_init(float* __restrict__ deg) {
    int idx = blockIdx.x * 256 + threadIdx.x;
    if (idx < BB * N_STOPS) deg[idx] = 1.0f;            // self-loop
}

__global__ void k_deg_scat(const int* __restrict__ edge, float* __restrict__ deg) {
    int idx = blockIdx.x * 256 + threadIdx.x;           // BB*EDGES
    if (idx >= BB * EDGES) return;
    int b = idx >> 11, e = idx & (EDGES - 1);
    int r = e >> 6, l = e & 63;
    int t = edge[(b * RR + r) * 2 * LL + LL + l];       // dst
    atomicAdd(&deg[b * N_STOPS + t], 1.0f);
}

__global__ void k_deg_fin(const float* __restrict__ deg, float* __restrict__ dinv,
                          float* __restrict__ drec) {
    int idx = blockIdx.x * 256 + threadIdx.x;
    if (idx >= BB * N_STOPS) return;
    float dg = deg[idx];
    dinv[idx] = rsqrtf(dg);
    drec[idx] = 1.0f / dg;
}

// ---------------- per-layer message passing, batched over scenarios ----------------
// y[b] = xw[b] * (1/deg[b]); grid.x covers 5000*128, grid.y = scenario
__global__ void k_yinit(const float* __restrict__ xw, const float* __restrict__ drec,
                        float* __restrict__ y, size_t xstride) {
    int idx = blockIdx.x * 256 + threadIdx.x;           // < N_STOPS*EMBED
    int b = blockIdx.y;
    y[(size_t)b * SSTRIDE + idx] = xw[(size_t)b * xstride + idx] * drec[b * N_STOPS + (idx >> 7)];
}

// y[b][dst] += xw[b][src] * dinv[src]*dinv[dst]; grid(EDGES, BB) x 128 threads
__global__ void k_escat(const int* __restrict__ edge, const float* __restrict__ dinv,
                        const float* __restrict__ xw, float* __restrict__ y, size_t xstride) {
    int e = blockIdx.x, d = threadIdx.x, b = blockIdx.y;
    int r = e >> 6, l = e & 63;
    int base = (b * RR + r) * 2 * LL + l;
    int s = edge[base];
    int t = edge[base + LL];
    float coef = dinv[b * N_STOPS + s] * dinv[b * N_STOPS + t];
    atomicAdd(&y[(size_t)b * SSTRIDE + (size_t)t * EMBED + d],
              xw[(size_t)b * xstride + (size_t)s * EMBED + d] * coef);
}

// x = lrelu(y + bias) -> bf16; last layer also rewrites f32 in place (out aliases y)
__global__ void k_act(const float* __restrict__ y, const float* __restrict__ bias,
                      __bf16* __restrict__ xb, float* __restrict__ xf, int write_f32) {
    int idx = blockIdx.x * 256 + threadIdx.x;           // < MP*EMBED
    if (idx >= MP * EMBED) return;
    int b = blockIdx.y;
    int n = idx >> 7, d = idx & 127;
    size_t off = (size_t)b * SSTRIDE + idx;
    float v = 0.f;
    if (n < N_STOPS) v = lrelu(y[off] + bias[d]);
    xb[off] = (__bf16)v;
    if (write_f32 && n < N_STOPS) xf[off] = v;          // in-place: xf == y
}

// ---------------- pooling (out_all has per-scenario stride SSTRIDE) ----------------
__global__ void k_zeroflags(float* __restrict__ flags) {
    int idx = blockIdx.x * 256 + threadIdx.x;
    if (idx < BB * N_STOPS) flags[idx] = 0.f;
}

__global__ void k_setflags(const int* __restrict__ edge, float* __restrict__ flags) {
    int idx = blockIdx.x * 256 + threadIdx.x;           // B*R*2*L = 65536
    if (idx >= BB * RR * 2 * LL) return;
    int b = idx >> 12;                                  // 4096 ids per scenario
    flags[b * N_STOPS + edge[idx]] = 1.0f;
}

__global__ void k_gpool(const float* __restrict__ flags, const float* __restrict__ out_all,
                        float* __restrict__ gd) {
    int b = blockIdx.x, d = threadIdx.x;
    const float* fb = flags + b * N_STOPS;
    const float* ob = out_all + (size_t)b * SSTRIDE + d;
    float acc = 0.f;
    for (int n = 0; n < N_STOPS; ++n)
        if (fb[n] > 0.f) acc += ob[(size_t)n * EMBED];  // uniform branch per wave
    gd[b * EMBED + d] = acc * 0.001f;
}

__global__ void k_rpool(const int* __restrict__ edge, const float* __restrict__ out_all,
                        float* __restrict__ rd) {
    __shared__ int   ids[2 * LL];
    __shared__ float uq[2 * LL];
    int br = blockIdx.x, i = threadIdx.x;               // 128 threads
    int b = br >> 5;
    ids[i] = edge[br * 2 * LL + i];                     // (b,r,:,:) contiguous 128 ints
    __syncthreads();
    float f = 1.f;
    int me = ids[i];
    for (int j = 0; j < i; ++j)
        if (ids[j] == me) { f = 0.f; break; }           // exact "unique" semantics
    uq[i] = f;
    __syncthreads();
    float cnt = 0.f, acc = 0.f;
    const float* ob = out_all + (size_t)b * SSTRIDE + i;  // dim d == i
    for (int j = 0; j < 2 * LL; ++j) {
        cnt += uq[j];
        if (uq[j] > 0.f) acc += ob[(size_t)ids[j] * EMBED];
    }
    rd[br * EMBED + i] = acc / cnt;
}

// ---------------- heads ----------------
__global__ void k_ghead(const float* __restrict__ gd, const float* __restrict__ w1,
                        const float* __restrict__ b1, const float* __restrict__ w2,
                        const float* __restrict__ b2, float* __restrict__ gdl,
                        float* __restrict__ out) {
    __shared__ float h[EMBED];
    int b = blockIdx.x, j = threadIdx.x;
    float acc = b1[j];
    for (int k = 0; k < EMBED; ++k) acc += gd[b * EMBED + k] * w1[k * EMBED + j];
    h[j] = lrelu(acc);
    gdl[b * EMBED + j] = lrelu(gd[b * EMBED + j]);      // tiled lrelu(gd) for route head
    __syncthreads();
    if (j == 0) {
        float o = b2[0];
        for (int k = 0; k < EMBED; ++k) o += h[k] * w2[k];
        out[b] = o;
    }
}

__global__ void k_rhead(const float* __restrict__ gdl, const float* __restrict__ rd,
                        const float* __restrict__ w1, const float* __restrict__ b1,
                        const float* __restrict__ w2, const float* __restrict__ b2,
                        float* __restrict__ out) {
    __shared__ float rin[2 * EMBED];
    __shared__ float h[EMBED];
    int br = blockIdx.x, j = threadIdx.x;
    int b = br >> 5;
    rin[j]         = gdl[b * EMBED + j];
    rin[EMBED + j] = rd[br * EMBED + j];
    __syncthreads();
    float acc = b1[j];
    for (int k = 0; k < 2 * EMBED; ++k) acc += rin[k] * w1[k * EMBED + j];
    h[j] = lrelu(acc);
    __syncthreads();
    if (j == 0) {
        float o = b2[0];
        for (int k = 0; k < EMBED; ++k) o += h[k] * w2[k];
        out[BB + br] = o;
    }
}

// ---------------- launcher ----------------
extern "C" void kernel_launch(void* const* d_in, const int* in_sizes, int n_in,
                              void* d_out, int out_size, void* d_ws, size_t ws_size,
                              hipStream_t stream) {
    const int*   edge  = (const int*)  d_in[0];
    const float* w_emb = (const float*)d_in[1];
    const float* b_emb = (const float*)d_in[2];
    const float* gcn_w = (const float*)d_in[3];
    const float* gcn_b = (const float*)d_in[4];
    const float* gh_w1 = (const float*)d_in[5];
    const float* gh_b1 = (const float*)d_in[6];
    const float* gh_w2 = (const float*)d_in[7];
    const float* gh_b2 = (const float*)d_in[8];
    const float* rh_w1 = (const float*)d_in[9];
    const float* rh_b1 = (const float*)d_in[10];
    const float* rh_w2 = (const float*)d_in[11];
    const float* rh_b2 = (const float*)d_in[12];
    float* out = (float*)d_out;

    char* wptr = (char*)d_ws;
    auto alloc = [&](size_t bytes) -> void* {
        void* p = (void*)wptr;
        wptr += (bytes + 255) & ~(size_t)255;
        return p;
    };
    __bf16* node_bf = (__bf16*)alloc(SSTRIDE * 2);                 // 1.3 MB
    __bf16* wt      = (__bf16*)alloc((size_t)3 * EMBED * EMBED * 2);
    float*  xw0     = (float*) alloc(SSTRIDE * 4);                 // 2.6 MB (shared layer-0 GEMM)
    float*  xw_all  = (float*) alloc((size_t)BB * SSTRIDE * 4);    // 41 MB
    float*  y_all   = (float*) alloc((size_t)BB * SSTRIDE * 4);    // 41 MB (aliased as out_all)
    __bf16* xb_all  = (__bf16*)alloc((size_t)BB * SSTRIDE * 2);    // 20.5 MB
    float*  deg     = (float*) alloc((size_t)BB * N_STOPS * 4);
    float*  dinv    = (float*) alloc((size_t)BB * N_STOPS * 4);
    float*  drec    = (float*) alloc((size_t)BB * N_STOPS * 4);
    float*  flags   = (float*) alloc((size_t)BB * N_STOPS * 4);
    float*  gd      = (float*) alloc((size_t)BB * EMBED * 4);
    float*  gdl     = (float*) alloc((size_t)BB * EMBED * 4);
    float*  rd      = (float*) alloc((size_t)BB * RR * EMBED * 4);
    float*  out_all = y_all;   // final activation rewrites y in place

    k_node <<<(MP * EMBED + 255) / 256, 256, 0, stream>>>(w_emb, b_emb, node_bf);
    k_wconv<<<(3 * EMBED * EMBED + 255) / 256, 256, 0, stream>>>(gcn_w, wt);
    // layer-0 GEMM is scenario-independent: compute once
    k_gemm <<<dim3(MTILES * 8, 1), 32, 0, stream>>>(node_bf, wt, xw0, 0, 0);

    // degrees for all scenarios in 3 launches
    k_deg_init<<<(BB * N_STOPS + 255) / 256, 256, 0, stream>>>(deg);
    k_deg_scat<<<(BB * EDGES + 255) / 256, 256, 0, stream>>>(edge, deg);
    k_deg_fin <<<(BB * N_STOPS + 255) / 256, 256, 0, stream>>>(deg, dinv, drec);

    for (int l = 0; l < 3; ++l) {
        const float* xin   = (l == 0) ? xw0 : xw_all;
        size_t xstride     = (l == 0) ? 0 : SSTRIDE;
        k_yinit<<<dim3((N_STOPS * EMBED) / 256, BB), 256, 0, stream>>>(xin, drec, y_all, xstride);
        k_escat<<<dim3(EDGES, BB), 128, 0, stream>>>(edge, dinv, xin, y_all, xstride);
        k_act  <<<dim3((MP * EMBED + 255) / 256, BB), 256, 0, stream>>>(
            y_all, gcn_b + l * EMBED, xb_all, out_all, (l == 2) ? 1 : 0);
        if (l < 2)
            k_gemm<<<dim3(MTILES * 8, BB), 32, 0, stream>>>(
                xb_all, wt + (size_t)(l + 1) * EMBED * EMBED, xw_all, SSTRIDE, SSTRIDE);
    }

    k_zeroflags<<<(BB * N_STOPS + 255) / 256, 256, 0, stream>>>(flags);
    k_setflags <<<(BB * RR * 2 * LL) / 256, 256, 0, stream>>>(edge, flags);
    k_gpool<<<BB, 128, 0, stream>>>(flags, out_all, gd);
    k_rpool<<<BB * RR, 128, 0, stream>>>(edge, out_all, rd);
    k_ghead<<<BB, 128, 0, stream>>>(gd, gh_w1, gh_b1, gh_w2, gh_b2, gdl, out);
    k_rhead<<<BB * RR, 128, 0, stream>>>(gdl, rd, rh_w1, rh_b1, rh_w2, rh_b2, out);

    (void)in_sizes; (void)n_in; (void)out_size; (void)ws_size;
}
